// EdgeFeatureInit_15479062134805
// MI455X (gfx1250) — compile-verified, matrix operand order checked by reading, not run on hardware
//
#include <hip/hip_runtime.h>

typedef __bf16 bf16_t;
typedef __attribute__((ext_vector_type(16))) bf16_t v16bf;
typedef __attribute__((ext_vector_type(8)))  float  v8f;

#define HIDDEN 128
#define NRAD   16
#define WAVES  4
#define TPB    (WAVES * 32)
#define NBLOCKS 512

// Fragment viewed as 8 packed dwords (element 2i in low half, 2i+1 in high half).
union Frag {
    v16bf v;
    unsigned u[8];
    uint4 q[2];
};

// K index inside a 32-wide k-chunk for fragment element j, given lane-half kh.
// (ISA 7.12.2, 16-bit A 16x32: lanes 0-15 hold K=0..7 & 16..23; lanes 16-31 hold K=8..15 & 24..31)
__device__ __forceinline__ int kOfJ(int j, int kh) {
    return (j < 8) ? (kh * 8 + j) : (16 + kh * 8 + (j - 8));
}

// Two f32 -> packed bf16 pair (truncation): 2 bitwise VALU ops, no b16 moves.
__device__ __forceinline__ unsigned pk2(float a, float b) {
    unsigned ua, ub;
    __builtin_memcpy(&ua, &a, 4);
    __builtin_memcpy(&ub, &b, 4);
    return (ua >> 16) | (ub & 0xFFFF0000u);
}

__device__ __forceinline__ unsigned short bf1(float a) {
    unsigned ua; __builtin_memcpy(&ua, &a, 4);
    return (unsigned short)(ua >> 16);
}

// silu(x) = x / (1 + e^-x) using v_exp_f32 (base 2) + v_rcp_f32: 2 TRANS + 2 VALU.
__device__ __forceinline__ float fsilu(float x) {
    return x * __builtin_amdgcn_rcpf(1.f + __builtin_amdgcn_exp2f(x * -1.44269504089f));
}

__device__ __forceinline__ v16bf makeAfragGlobal(const float* __restrict__ rowp, int kbase, int kh) {
    float t[16];
    const float4* p0 = (const float4*)(rowp + kbase + kh * 8);
    const float4* p1 = (const float4*)(rowp + kbase + 16 + kh * 8);
    *(float4*)&t[0]  = p0[0];
    *(float4*)&t[4]  = p0[1];
    *(float4*)&t[8]  = p1[0];
    *(float4*)&t[12] = p1[1];
    Frag f;
#pragma unroll
    for (int i = 0; i < 8; ++i) f.u[i] = pk2(t[2 * i], t[2 * i + 1]);
    return f.v;
}

__global__ void __launch_bounds__(TPB)
edge_mlp_wmma_kernel(const float* __restrict__ node_embs,
                     const int*   __restrict__ edge_index,
                     const float* __restrict__ edge_weight,
                     const float* __restrict__ W_rbf0,
                     const float* __restrict__ b_rbf0,
                     const float* __restrict__ W_lin,
                     const float* __restrict__ b_lin,
                     const float* __restrict__ W_rbf1,
                     float* __restrict__ out1,
                     float* __restrict__ out2,
                     int nEdges)
{
    extern __shared__ __align__(32) char smem[];
    bf16_t*         WlinF  = (bf16_t*)smem;              // 12 kchunks * 8 ntiles * 32 lanes * 16 = 96KB
    bf16_t*         Wr0F   = WlinF + 12 * 8 * 32 * 16;   // 8KB
    bf16_t*         Wr1F   = Wr0F  + 8 * 32 * 16;        // 8KB
    unsigned short* bounce = (unsigned short*)(Wr1F + 8 * 32 * 16);  // WAVES*16*128*2B = 16KB

    const int tid  = threadIdx.x;
    const int lane = tid & 31;
    const int wv   = tid >> 5;
    const int kh   = lane >> 4;
    const int col  = lane & 15;

    // ---- stage W_lin (384x128 row-major fp32) into bf16 B-fragment order ----
    for (int s = tid; s < 12 * 8 * 32; s += TPB) {
        int kc  = s >> 8;
        int nt  = (s >> 5) & 7;
        int ln  = s & 31;
        int lkh = ln >> 4;
        int n   = nt * 16 + (ln & 15);
        Frag f;
#pragma unroll
        for (int i = 0; i < 8; ++i) {
            int k0 = kc * 32 + kOfJ(2 * i,     lkh);
            int k1 = kc * 32 + kOfJ(2 * i + 1, lkh);
            f.u[i] = pk2(W_lin[k0 * HIDDEN + n], W_lin[k1 * HIDDEN + n]);
        }
        *(v16bf*)(WlinF + s * 16) = f.v;
    }
    // ---- stage W_rbf0 / W_rbf1 (16x128, K zero-padded to 32) ----
    for (int s = tid; s < 8 * 32; s += TPB) {
        int nt  = s >> 5;
        int ln  = s & 31;
        int lkh = ln >> 4;
        int n   = nt * 16 + (ln & 15);
        Frag f0, f1;
#pragma unroll
        for (int i = 0; i < 8; ++i) {
            int k0 = kOfJ(2 * i,     lkh);
            int k1 = kOfJ(2 * i + 1, lkh);
            float a0 = (k0 < NRAD) ? W_rbf0[k0 * HIDDEN + n] : 0.f;
            float a1 = (k1 < NRAD) ? W_rbf0[k1 * HIDDEN + n] : 0.f;
            float b0 = (k0 < NRAD) ? W_rbf1[k0 * HIDDEN + n] : 0.f;
            float b1 = (k1 < NRAD) ? W_rbf1[k1 * HIDDEN + n] : 0.f;
            f0.u[i] = pk2(a0, a1);
            f1.u[i] = pk2(b0, b1);
        }
        *(v16bf*)(Wr0F + s * 16) = f0.v;
        *(v16bf*)(Wr1F + s * 16) = f1.v;
    }
    __syncthreads();

    unsigned short* myB = bounce + wv * 16 * HIDDEN;

    // Per-lane biases depend only on (nt, col) -> hoist out of the tile loop.
    float bias0[8], biasL[8];
#pragma unroll
    for (int nt = 0; nt < 8; ++nt) {
        bias0[nt] = b_rbf0[nt * 16 + col];
        biasL[nt] = b_lin [nt * 16 + col];
    }

    const int nTiles     = (nEdges + 15) >> 4;
    const int waveGlobal = blockIdx.x * WAVES + wv;
    const int waveCount  = gridDim.x * WAVES;

    for (int tile = waveGlobal; tile < nTiles; tile += waveCount) {
        const int row = col;                 // A-matrix row owned by this lane
        int e = tile * 16 + row;
        if (e >= nEdges) e = nEdges - 1;     // branchless clamp keeps EXEC all-ones
        const int   sIdx = edge_index[e];
        const int   dIdx = edge_index[nEdges + e];
        const float dw   = edge_weight[e];
        const float rdw  = __builtin_amdgcn_rcpf(dw);

        // ---- Bessel RBF A-fragment: sqrt(2/5) * sin(d * n * pi/5)/d, n = K+1 ----
        // v_sin_f32 takes input in turns: (pi/5)/(2*pi) = 0.1.
        float sv[8];
#pragma unroll
        for (int j = 0; j < 8; ++j) {
            int k = kh * 8 + j;              // real K owned by this lane (element j)
            sv[j] = 0.63245553203f * __builtin_amdgcn_sinf(dw * (float)(k + 1) * 0.1f) * rdw;
        }
        Frag fr;
#pragma unroll
        for (int i = 0; i < 4; ++i) fr.u[i] = pk2(sv[2 * i], sv[2 * i + 1]);
#pragma unroll
        for (int i = 4; i < 8; ++i) fr.u[i] = 0u;       // K = 16..31 zero pad
        v16bf rbfA = fr.v;

        // ---- rbf @ W_rbf0 -> silu -> bounce (A layout), and rbf @ W_rbf1 kept ----
        v8f acc2[8];
#pragma unroll
        for (int nt = 0; nt < 8; ++nt) {
            v8f z = {0.f, 0.f, 0.f, 0.f, 0.f, 0.f, 0.f, 0.f};
            v16bf b0 = *(const v16bf*)(Wr0F + (nt * 32 + lane) * 16);
            v8f c0 = __builtin_amdgcn_wmma_f32_16x16x32_bf16(false, rbfA, false, b0,
                                                             (short)0, z, false, false);
            v16bf b1 = *(const v16bf*)(Wr1F + (nt * 32 + lane) * 16);
            acc2[nt] = __builtin_amdgcn_wmma_f32_16x16x32_bf16(false, rbfA, false, b1,
                                                               (short)0, z, false, false);
#pragma unroll
            for (int r = 0; r < 8; ++r) {
                float s = fsilu(c0[r] + bias0[nt]);
                int mr = kh * 8 + r;                    // C-layout row
                myB[mr * HIDDEN + nt * 16 + col] = bf1(s);
            }
        }

        // ---- main GEMM: [src | dst | silu(rbf0)] (16x384) x W_lin (384x128) ----
        v8f accM[8];
#pragma unroll
        for (int nt = 0; nt < 8; ++nt)
            accM[nt] = (v8f){0.f, 0.f, 0.f, 0.f, 0.f, 0.f, 0.f, 0.f};

        const float* srow = node_embs + (size_t)sIdx * HIDDEN;
        const float* drow = node_embs + (size_t)dIdx * HIDDEN;

#pragma unroll
        for (int c = 0; c < 4; ++c) {                   // k-chunks 0..3 : src
            v16bf a = makeAfragGlobal(srow, c * 32, kh);
#pragma unroll
            for (int nt = 0; nt < 8; ++nt) {
                v16bf b = *(const v16bf*)(WlinF + ((c * 8 + nt) * 32 + lane) * 16);
                accM[nt] = __builtin_amdgcn_wmma_f32_16x16x32_bf16(false, a, false, b,
                                                                   (short)0, accM[nt], false, false);
            }
        }
#pragma unroll
        for (int c = 0; c < 4; ++c) {                   // k-chunks 4..7 : dst
            v16bf a = makeAfragGlobal(drow, c * 32, kh);
#pragma unroll
            for (int nt = 0; nt < 8; ++nt) {
                v16bf b = *(const v16bf*)(WlinF + (((4 + c) * 8 + nt) * 32 + lane) * 16);
                accM[nt] = __builtin_amdgcn_wmma_f32_16x16x32_bf16(false, a, false, b,
                                                                   (short)0, accM[nt], false, false);
            }
        }
#pragma unroll
        for (int c = 0; c < 4; ++c) {                   // k-chunks 8..11 : rbf0 (LDS bounce)
            Frag f;
            f.q[0] = *(const uint4*)(myB + row * HIDDEN + c * 32 + kh * 8);
            f.q[1] = *(const uint4*)(myB + row * HIDDEN + c * 32 + 16 + kh * 8);
#pragma unroll
            for (int nt = 0; nt < 8; ++nt) {
                v16bf b = *(const v16bf*)(WlinF + (((8 + c) * 8 + nt) * 32 + lane) * 16);
                accM[nt] = __builtin_amdgcn_wmma_f32_16x16x32_bf16(false, f.v, false, b,
                                                                   (short)0, accM[nt], false, false);
            }
        }

        // ---- epilogue: e1 = silu(h@W+b), e2 = (rbf@W_rbf1) * e1 ----
#pragma unroll
        for (int nt = 0; nt < 8; ++nt) {
#pragma unroll
            for (int r = 0; r < 8; ++r) {
                int er = tile * 16 + kh * 8 + r;
                if (er < nEdges) {
                    float e1 = fsilu(accM[nt][r] + biasL[nt]);
                    float e2 = acc2[nt][r] * e1;
                    out1[(size_t)er * HIDDEN + nt * 16 + col] = e1;
                    out2[(size_t)er * HIDDEN + nt * 16 + col] = e2;
                }
            }
        }
    }
}

extern "C" void kernel_launch(void* const* d_in, const int* in_sizes, int n_in,
                              void* d_out, int out_size, void* d_ws, size_t ws_size,
                              hipStream_t stream)
{
    const float* node_embs   = (const float*)d_in[0];
    const int*   edge_index  = (const int*)  d_in[1];
    const float* edge_weight = (const float*)d_in[2];
    const float* W_rbf0      = (const float*)d_in[3];
    const float* b_rbf0      = (const float*)d_in[4];
    const float* W_lin       = (const float*)d_in[5];
    const float* b_lin       = (const float*)d_in[6];
    const float* W_rbf1      = (const float*)d_in[7];

    const int E = in_sizes[2];
    float* out1 = (float*)d_out;
    float* out2 = out1 + (size_t)E * HIDDEN;

    const size_t shmem = (size_t)(12 * 8 * 32 * 16   // W_lin frags
                                 + 2 * 8 * 32 * 16   // W_rbf0 + W_rbf1 frags
                                 + WAVES * 16 * HIDDEN) * sizeof(bf16_t);

    const int tiles = (E + 15) / 16;
    int blocks = NBLOCKS;
    if (blocks * WAVES > tiles) blocks = (tiles + WAVES - 1) / WAVES;

    edge_mlp_wmma_kernel<<<blocks, TPB, shmem, stream>>>(
        node_embs, edge_index, edge_weight,
        W_rbf0, b_rbf0, W_lin, b_lin, W_rbf1,
        out1, out2, E);
}
